// ModelLSTM_31172872634510
// MI455X (gfx1250) — matrix operand order repaired and checked by reading, not compiled, not run on hardware
//
#include <hip/hip_runtime.h>
#include <stdint.h>

// ---------------- CDNA5 WMMA types/helpers ----------------
typedef __attribute__((ext_vector_type(16))) __bf16 v16bf;
typedef __attribute__((ext_vector_type(8)))  float  v8f;
typedef int v4i_vec __attribute__((vector_size(16)));   // matches builtin param type

union FragB16 { v16bf v; uint4 q[2]; };
union AccU    { v8f v; float f[8]; };

static __device__ __forceinline__ uint16_t f2bf(float f) {
    uint32_t u = __float_as_uint(f);
    uint32_t r = (u + 0x7FFFu + ((u >> 16) & 1u)) >> 16;   // RNE
    return (uint16_t)r;
}

static __device__ __forceinline__ v8f wmma_bf16(const FragB16& a, const FragB16& b, v8f c) {
    return __builtin_amdgcn_wmma_f32_16x16x32_bf16(false, a.v, false, b.v, (short)0, c, false, false);
}

// ---- async global->LDS copy (CDNA5 GLOBAL_LOAD_ASYNC_TO_LDS_B128, ASYNCcnt) ----
#if defined(__gfx1250__) && __has_builtin(__builtin_amdgcn_global_load_async_to_lds_b128) && \
    __has_builtin(__builtin_amdgcn_s_wait_asynccnt)
#define HAVE_ASYNC_LDS 1
#else
#define HAVE_ASYNC_LDS 0
#endif

static __device__ __forceinline__ void async_copy16(void* lds_dst, const void* gsrc) {
#if HAVE_ASYNC_LDS
    __builtin_amdgcn_global_load_async_to_lds_b128(
        (__attribute__((address_space(1))) v4i_vec*)gsrc,
        (__attribute__((address_space(3))) v4i_vec*)lds_dst,
        0, 0);
#else
    *(uint4*)lds_dst = *(const uint4*)gsrc;
#endif
}

static __device__ __forceinline__ void async_wait_all() {
#if HAVE_ASYNC_LDS
    __builtin_amdgcn_s_wait_asynccnt(0);
#endif
}

// A-matrix 16x32 bf16 fragment from LDS, row-major A[m][k], row stride Kld (halves).
// ISA layout: lane<16 holds K {0..7,16..23}, lane>=16 holds K {8..15,24..31}, row M = lane&15.
static __device__ __forceinline__ FragB16 load_a_lds(const uint16_t* base, int kc, int Kld) {
    int lane = threadIdx.x & 31;
    int m = lane & 15, hi = lane >> 4;
    const uint16_t* p = base + m * Kld + kc * 32 + hi * 8;
    FragB16 f;
    f.q[0] = *(const uint4*)(p);
    f.q[1] = *(const uint4*)(p + 16);
    return f;
}

// B-matrix 32x16 bf16 fragment from global weights W[N][K] row-major (we compute A*W^T).
// ISA layout: lane holds column N = lane&15; lane<16 K {0..7,16..23}, lane>=16 K {8..15,24..31} of the chunk.
static __device__ __forceinline__ FragB16 load_b_glb(const uint16_t* W, int n0, int kc, int K) {
    int lane = threadIdx.x & 31;
    int n = n0 + (lane & 15), hi = lane >> 4;
    const uint16_t* p = W + (size_t)n * K + kc * 32 + hi * 8;
    FragB16 f;
    f.q[0] = *(const uint4*)(p);
    f.q[1] = *(const uint4*)(p + 16);
    return f;
}

static __device__ __forceinline__ float sigf(float x) { return 1.0f / (1.0f + expf(-x)); }

// ---------------- small utility kernels ----------------
__global__ void k_zero(float* p, long n) {
    long i = (long)blockIdx.x * blockDim.x + threadIdx.x;
    if (i < n) p[i] = 0.f;
}

__global__ void k_f2bf(const float* __restrict__ s, uint16_t* __restrict__ d, long n) {
    long i = (long)blockIdx.x * blockDim.x + threadIdx.x;
    if (i < n) d[i] = f2bf(s[i]);
}

// pad d_w (100x512) into 128x512 bf16 (zeros beyond row 99)
__global__ void k_dw_pad(const float* __restrict__ dw, uint16_t* __restrict__ dst) {
    int i = blockIdx.x * blockDim.x + threadIdx.x;   // 0..65535
    int r = i >> 9, c = i & 511;
    dst[i] = (r < 100) ? f2bf(dw[r * 512 + c]) : (uint16_t)0;
}

// ---------------- embedding gather: xb[b][t][d] = bf16(emb[sents[b,t]][d]) ----------------
__global__ void k_embed(const int* __restrict__ sents, const float* __restrict__ emb,
                        uint16_t* __restrict__ xb) {
    int row = blockIdx.x;                 // b*T + t
    int tok = sents[row];
    const float* src = emb + (size_t)tok * 512;
    uint16_t* dst = xb + (size_t)row * 512;
    for (int d = threadIdx.x; d < 512; d += blockDim.x) dst[d] = f2bf(src[d]);
}

// ---------------- input-side gate GEMM: gx[dir][t][b][g] = x[b][tt] @ Wih^T + bih + bhh ----------
// grid (T, 2dirs), block 256 (8 waves). M-tile = all 16 batches of one timestep.
__global__ void k_gate_gemm(const uint16_t* __restrict__ xin, const uint16_t* __restrict__ Wihb,
                            const float* __restrict__ bih, const float* __restrict__ bhh,
                            float* __restrict__ gx, int layer) {
    __shared__ __align__(16) uint16_t As[16 * 512];   // 16KB A tile
    int t = blockIdx.x, dir = blockIdx.y;
    int tt = dir ? (1023 - t) : t;
    // stage A: 16 rows (b) x 512 bf16, via async global->LDS when available
    for (int q = 0; q < 4; q++) {
        int u = threadIdx.x + q * 256;          // 1024 uint4 chunks
        int m = u >> 6, c = u & 63;
        async_copy16(As + m * 512 + c * 8,
                     xin + ((size_t)m * 1024 + tt) * 512 + c * 8);
    }
    async_wait_all();
    __syncthreads();
    const uint16_t* W = Wihb + (size_t)(layer * 2 + dir) * 1024 * 512;
    int wv = threadIdx.x >> 5;
    v8f acc[8] = {};
    for (int kc = 0; kc < 16; kc++) {
        FragB16 a = load_a_lds(As, kc, 512);
        FragB16 bf[8];
#pragma unroll
        for (int tI = 0; tI < 8; tI++) {
            int n0 = (wv * 8 + tI) * 16;
            __builtin_prefetch(W + (size_t)n0 * 512 + (kc + 1) * 32, 0, 1);
            bf[tI] = load_b_glb(W, n0, kc, 512);
        }
#pragma unroll
        for (int tI = 0; tI < 8; tI++)
            acc[tI] = wmma_bf16(a, bf[tI], acc[tI]);
    }
    int lane = threadIdx.x & 31, hi = lane >> 4, nn = lane & 15;
    const float* bi = bih + (layer * 2 + dir) * 1024;
    const float* bh = bhh + (layer * 2 + dir) * 1024;
    float* gout = gx + ((size_t)dir * 1024 + t) * 16 * 1024;
#pragma unroll
    for (int tI = 0; tI < 8; tI++) {
        int n = (wv * 8 + tI) * 16 + nn;
        float bb = bi[n] + bh[n];
        AccU u; u.v = acc[tI];
#pragma unroll
        for (int r = 0; r < 8; r++) {
            int m = r + hi * 8;
            gout[m * 1024 + n] = u.f[r] + bb;
        }
    }
}

// ---------------- sequential recurrence: one 32-wave workgroup per direction ----------------
// Whh fragments live in VGPRs for all 1024 steps; h in LDS (bf16), c in registers.
__global__ void k_lstm_rec(const float* __restrict__ gx, const uint16_t* __restrict__ Whhb,
                           uint16_t* __restrict__ hout_bf, float* __restrict__ hout_f32,
                           int layer, int final_layer) {
    __shared__ __align__(16) float    gmat[16 * 1024];   // 64KB gate pre-activations
    __shared__ __align__(16) uint16_t hls[16 * 256];     // 8KB  h (bf16)
    int dir = blockIdx.x;
    const uint16_t* W = Whhb + (size_t)(layer * 2 + dir) * 1024 * 256;
    const float* gxd = gx + (size_t)dir * 1024 * 16 * 1024;
    int tid = threadIdx.x;
    int wv = tid >> 5, lane = tid & 31, hi = lane >> 4, nn = lane & 15;
    // wave wv owns gate columns [32*wv, 32*wv+32): 2 N-tiles. Preload Whh fragments.
    FragB16 Bf[2][8];
#pragma unroll
    for (int tI = 0; tI < 2; tI++)
#pragma unroll
        for (int kc = 0; kc < 8; kc++)
            Bf[tI][kc] = load_b_glb(W, (wv * 2 + tI) * 16, kc, 256);
    for (int u = tid; u < 16 * 256; u += 1024) hls[u] = 0;
    float c[4] = {0.f, 0.f, 0.f, 0.f};
    int hidx = tid & 255, bgrp = tid >> 8;
    __syncthreads();
    for (int t = 0; t < 1024; t++) {
        // h(16x256) @ Whh^T(256x1024), this wave's 32 columns
        v8f acc0 = {}, acc1 = {};
#pragma unroll
        for (int kc = 0; kc < 8; kc++) {
            FragB16 a = load_a_lds(hls, kc, 256);
            acc0 = wmma_bf16(a, Bf[0][kc], acc0);
            acc1 = wmma_bf16(a, Bf[1][kc], acc1);
        }
        const float* gt = gxd + (size_t)t * 16 * 1024;
#pragma unroll
        for (int tI = 0; tI < 2; tI++) {
            int n = (wv * 2 + tI) * 16 + nn;
            AccU u; u.v = tI ? acc1 : acc0;
#pragma unroll
            for (int r = 0; r < 8; r++) {
                int m = r + hi * 8;
                gmat[m * 1024 + n] = u.f[r] + gt[m * 1024 + n];
            }
        }
        __syncthreads();
        // elementwise LSTM cell; each thread owns 4 (b,hidx) cells across all steps
        int tt = dir ? (1023 - t) : t;
#pragma unroll
        for (int k = 0; k < 4; k++) {
            int b = bgrp * 4 + k;
            float gi = gmat[b * 1024 + hidx];
            float gf = gmat[b * 1024 + 256 + hidx];
            float gg = gmat[b * 1024 + 512 + hidx];
            float go = gmat[b * 1024 + 768 + hidx];
            float cn = sigf(gf) * c[k] + sigf(gi) * tanhf(gg);
            float h = sigf(go) * tanhf(cn);
            c[k] = cn;
            hls[b * 256 + hidx] = f2bf(h);
            size_t oidx = ((size_t)b * 1024 + tt) * 512 + dir * 256 + hidx;
            if (final_layer) hout_f32[oidx] = h;
            else             hout_bf[oidx] = f2bf(h);
        }
        __syncthreads();
    }
}

// ---------------- segment means + scatter-add into entity slots ----------------
__global__ void k_segments(const int* __restrict__ ent, const float* __restrict__ x,
                           float* __restrict__ cont, float* __restrict__ cmask) {
    int b = blockIdx.y, e = blockIdx.x;
    const int* row = ent + (b * 64 + e) * 3;
    int id = row[0], st = row[1], en = row[2];
    if (id < 0) return;
    int idc = min(max(id, 0), 63);
    int len = en - st; if (len < 1) len = 1;
    float inv = 1.0f / (float)len;
    for (int d = threadIdx.x; d < 512; d += blockDim.x) {
        float s = 0.f;
        for (int t = st; t < en; t++) s += x[((size_t)b * 1024 + t) * 512 + d];
        atomicAdd(&cont[((size_t)b * 64 + idc) * 512 + d], s * inv);
    }
    if (threadIdx.x == 0) atomicAdd(&cmask[b * 64 + idc], 1.0f);
}

// ---------------- layernorm over D=512 + bf16 convert ----------------
__global__ void k_layernorm(const float* __restrict__ cont, const float* __restrict__ g,
                            const float* __restrict__ bta, uint16_t* __restrict__ contb) {
    __shared__ float red[256];
    int r = blockIdx.x, tid = threadIdx.x;
    const float* src = cont + (size_t)r * 512;
    float v0 = src[tid], v1 = src[tid + 256];
    red[tid] = v0 + v1; __syncthreads();
    for (int s = 128; s > 0; s >>= 1) { if (tid < s) red[tid] += red[tid + s]; __syncthreads(); }
    float mu = red[0] * (1.0f / 512.0f); __syncthreads();
    float d0 = v0 - mu, d1 = v1 - mu;
    red[tid] = d0 * d0 + d1 * d1; __syncthreads();
    for (int s = 128; s > 0; s >>= 1) { if (tid < s) red[tid] += red[tid + s]; __syncthreads(); }
    float rstd = rsqrtf(red[0] * (1.0f / 512.0f) + 1e-5f);
    uint16_t* dst = contb + (size_t)r * 512;
    dst[tid]       = f2bf(d0 * rstd * g[tid] + bta[tid]);
    dst[tid + 256] = f2bf(d1 * rstd * g[tid + 256] + bta[tid + 256]);
}

// ---------------- rel1/rel2 projections: (B*E,512) @ (512,512)^T ----------------
__global__ void k_rel_gemm(const uint16_t* __restrict__ contb,
                           const uint16_t* __restrict__ w1, const float* __restrict__ b1,
                           const uint16_t* __restrict__ w2, const float* __restrict__ b2,
                           float* __restrict__ rel1, float* __restrict__ rel2) {
    __shared__ __align__(16) uint16_t As[16 * 512];
    int mt = blockIdx.x, which = blockIdx.y;
    const uint16_t* W = which ? w2 : w1;
    const float* bias = which ? b2 : b1;
    float* out = which ? rel2 : rel1;
    for (int q = 0; q < 4; q++) {
        int u = threadIdx.x + q * 256;
        int m = u >> 6, cc = u & 63;
        async_copy16(As + m * 512 + cc * 8,
                     contb + ((size_t)(mt * 16 + m)) * 512 + cc * 8);
    }
    async_wait_all();
    __syncthreads();
    int wv = threadIdx.x >> 5, lane = threadIdx.x & 31, hi = lane >> 4, nn = lane & 15;
    v8f acc[4] = {};
    for (int kc = 0; kc < 16; kc++) {
        FragB16 a = load_a_lds(As, kc, 512);
        FragB16 bf[4];
#pragma unroll
        for (int tI = 0; tI < 4; tI++)
            bf[tI] = load_b_glb(W, (wv * 4 + tI) * 16, kc, 512);
#pragma unroll
        for (int tI = 0; tI < 4; tI++)
            acc[tI] = wmma_bf16(a, bf[tI], acc[tI]);
    }
#pragma unroll
    for (int tI = 0; tI < 4; tI++) {
        int n = (wv * 4 + tI) * 16 + nn;
        float bb = bias[n];
        AccU u; u.v = acc[tI];
#pragma unroll
        for (int r = 0; r < 8; r++) {
            int m = r + hi * 8;
            out[((size_t)(mt * 16 + m)) * 512 + n] = u.f[r] + bb;
        }
    }
}

// ---------------- fused pairwise MLP + head + mask + log_softmax ----------------
// grid (E=64 j, E/16=4 itiles, B=16), block 256 (8 waves).
__global__ void k_pairwise(const float* __restrict__ rel1, const float* __restrict__ rel2,
                           const uint16_t* __restrict__ pwb, const float* __restrict__ p_b,
                           const uint16_t* __restrict__ dwb, const float* __restrict__ d_b,
                           const float* __restrict__ cmask, float* __restrict__ outp) {
    __shared__ __align__(16) uint16_t Ts[16 * 512];  // relu(rel1_i + rel2_j), bf16
    __shared__ __align__(16) uint16_t Hs[16 * 512];  // relu(T@p_w^T + p_b), bf16
    __shared__ float Ys[16 * 128];
    int j = blockIdx.x, itile = blockIdx.y, b = blockIdx.z;
    int i0 = itile * 16;
    const float* r2row = rel2 + ((size_t)b * 64 + j) * 512;
    for (int q = 0; q < 32; q++) {
        int u = threadIdx.x + q * 256;       // 8192 elements
        int m = u >> 9, d = u & 511;
        float v = rel1[((size_t)b * 64 + i0 + m) * 512 + d] + r2row[d];
        Ts[m * 512 + d] = f2bf(v > 0.f ? v : 0.f);
    }
    __syncthreads();
    int wv = threadIdx.x >> 5, lane = threadIdx.x & 31, hi = lane >> 4, nn = lane & 15;
    v8f acc[4] = {};
    for (int kc = 0; kc < 16; kc++) {
        FragB16 a = load_a_lds(Ts, kc, 512);
        FragB16 bf[4];
#pragma unroll
        for (int tI = 0; tI < 4; tI++)
            bf[tI] = load_b_glb(pwb, (wv * 4 + tI) * 16, kc, 512);
#pragma unroll
        for (int tI = 0; tI < 4; tI++)
            acc[tI] = wmma_bf16(a, bf[tI], acc[tI]);
    }
#pragma unroll
    for (int tI = 0; tI < 4; tI++) {
        int n = (wv * 4 + tI) * 16 + nn;
        float bb = p_b[n];
        AccU u; u.v = acc[tI];
#pragma unroll
        for (int r = 0; r < 8; r++) {
            int m = r + hi * 8;
            float h = u.f[r] + bb;
            Hs[m * 512 + n] = f2bf(h > 0.f ? h : 0.f);
        }
    }
    __syncthreads();
    // head GEMM: each wave one 16-col tile of 128 (d_w zero-padded past row 99)
    v8f acc2 = {};
    for (int kc = 0; kc < 16; kc++) {
        FragB16 a = load_a_lds(Hs, kc, 512);
        FragB16 bfr = load_b_glb(dwb, wv * 16, kc, 512);
        acc2 = wmma_bf16(a, bfr, acc2);
    }
    {
        int n = wv * 16 + nn;
        float bb = (n < 100) ? d_b[n] : 0.f;
        AccU u; u.v = acc2;
        float cmj = fminf(cmask[b * 64 + j], 1.f);
#pragma unroll
        for (int r = 0; r < 8; r++) {
            int m = r + hi * 8;
            float cm = fminf(cmask[b * 64 + i0 + m], 1.f) * cmj;
            Ys[m * 128 + n] = (u.f[r] + bb) * cm;
        }
    }
    __syncthreads();
    if (threadIdx.x < 16) {
        int m = threadIdx.x;
        float mx = -1e30f;
        for (int n = 0; n < 100; n++) mx = fmaxf(mx, Ys[m * 128 + n]);
        float s = 0.f;
        for (int n = 0; n < 100; n++) s += expf(Ys[m * 128 + n] - mx);
        float lz = logf(s) + mx;
        float* o = outp + (((size_t)(b * 64 + j) * 64) + (i0 + m)) * 100;
        for (int n = 0; n < 100; n++) o[n] = Ys[m * 128 + n] - lz;
    }
}

// ---------------- host-side orchestration ----------------
extern "C" void kernel_launch(void* const* d_in, const int* in_sizes, int n_in,
                              void* d_out, int out_size, void* d_ws, size_t ws_size,
                              hipStream_t stream) {
    (void)in_sizes; (void)n_in; (void)out_size; (void)ws_size;
    const int*   sents = (const int*)d_in[0];
    const int*   ent   = (const int*)d_in[1];
    const float* emb   = (const float*)d_in[3];
    const float* Wih   = (const float*)d_in[4];
    const float* Whh   = (const float*)d_in[5];
    const float* bih   = (const float*)d_in[6];
    const float* bhh   = (const float*)d_in[7];
    const float* ln_g  = (const float*)d_in[8];
    const float* ln_b  = (const float*)d_in[9];
    const float* r1w   = (const float*)d_in[10];
    const float* r1b   = (const float*)d_in[11];
    const float* r2w   = (const float*)d_in[12];
    const float* r2b   = (const float*)d_in[13];
    const float* pw    = (const float*)d_in[14];
    const float* pb    = (const float*)d_in[15];
    const float* dw    = (const float*)d_in[16];
    const float* db    = (const float*)d_in[17];
    float* outp = (float*)d_out;

    char* ws = (char*)d_ws;
    size_t off = 0;
    auto alloc = [&](size_t bytes) -> char* {
        char* p = ws + off;
        off += (bytes + 255) & ~(size_t)255;
        return p;
    };
    uint16_t* xb    = (uint16_t*)alloc(16ull * 1024 * 512 * 2);   // layer0 input, bf16
    uint16_t* hbA   = (uint16_t*)alloc(16ull * 1024 * 512 * 2);   // layer0 output, bf16
    float*    xf32  = (float*)   alloc(16ull * 1024 * 512 * 4);   // layer1 output, f32
    float*    gx    = (float*)   alloc(2ull * 1024 * 16 * 1024 * 4); // gates per dir
    uint16_t* Wihb  = (uint16_t*)alloc(2ull * 2 * 1024 * 512 * 2);
    uint16_t* Whhb  = (uint16_t*)alloc(2ull * 2 * 1024 * 256 * 2);
    uint16_t* r1wb  = (uint16_t*)alloc(512ull * 512 * 2);
    uint16_t* r2wb  = (uint16_t*)alloc(512ull * 512 * 2);
    uint16_t* pwb   = (uint16_t*)alloc(512ull * 512 * 2);
    uint16_t* dwb   = (uint16_t*)alloc(128ull * 512 * 2);
    float*    cont  = (float*)   alloc(16ull * 64 * 512 * 4);
    float*    cmask = (float*)   alloc(16ull * 64 * 4);
    uint16_t* contb = (uint16_t*)alloc(16ull * 64 * 512 * 2);
    float*    rel1  = (float*)   alloc(16ull * 64 * 512 * 4);
    float*    rel2  = (float*)   alloc(16ull * 64 * 512 * 4);

    // weight conversions (every call; deterministic)
    k_f2bf<<<(2097152 + 255) / 256, 256, 0, stream>>>(Wih, Wihb, 2097152);
    k_f2bf<<<(1048576 + 255) / 256, 256, 0, stream>>>(Whh, Whhb, 1048576);
    k_f2bf<<<(262144 + 255) / 256, 256, 0, stream>>>(r1w, r1wb, 262144);
    k_f2bf<<<(262144 + 255) / 256, 256, 0, stream>>>(r2w, r2wb, 262144);
    k_f2bf<<<(262144 + 255) / 256, 256, 0, stream>>>(pw, pwb, 262144);
    k_dw_pad<<<(65536 + 255) / 256, 256, 0, stream>>>(dw, dwb);
    k_zero<<<(524288 + 255) / 256, 256, 0, stream>>>(cont, 524288);
    k_zero<<<(1024 + 255) / 256, 256, 0, stream>>>(cmask, 1024);

    k_embed<<<16384, 256, 0, stream>>>(sents, emb, xb);

    for (int l = 0; l < 2; l++) {
        const uint16_t* xin = (l == 0) ? xb : hbA;
        k_gate_gemm<<<dim3(1024, 2), 256, 0, stream>>>(xin, Wihb, bih, bhh, gx, l);
        k_lstm_rec<<<2, 1024, 0, stream>>>(gx, Whhb, hbA, xf32, l, l == 1);
    }

    k_segments<<<dim3(64, 16), 256, 0, stream>>>(ent, xf32, cont, cmask);
    k_layernorm<<<1024, 256, 0, stream>>>(cont, ln_g, ln_b, contb);
    k_rel_gemm<<<dim3(64, 2), 256, 0, stream>>>(contb, r1wb, r1b, r2wb, r2b, rel1, rel2);
    k_pairwise<<<dim3(64, 4, 16), 256, 0, stream>>>(rel1, rel2, pwb, pb, dwb, db, cmask, outp);
}
